// Hgru2_1d_8521215115682
// MI455X (gfx1250) — matrix operand order repaired and checked by reading, compiled, not run on hardware
//
#include <hip/hip_runtime.h>
#include <hip/hip_bf16.h>

// ---------------------------------------------------------------------------
// HGRU2 layer for MI455X (gfx1250), bf16 WMMA path.
//   feat = x @ W_in + b_in ; gates ; diagonal scan ; gated readout ; LN ;
//   out = LN(o) @ W_out + b_out
// GEMMs: v_wmma_f32_16x16x32_bf16, 2x2 register tiling per wave (32x32 C),
// double-buffered fragments so WMMA overlaps the next fragment loads.
// ---------------------------------------------------------------------------

typedef __attribute__((ext_vector_type(16))) __bf16 v16bf;
typedef __attribute__((ext_vector_type(8)))  __bf16 v8bf;
typedef __attribute__((ext_vector_type(8)))  float  v8f;

#define SEQ   2048
#define BATCH 4
#define DIM   1024
#define ROWS  (SEQ * BATCH)       // 8192 flattened (n,b) rows
#define N3    (3 * DIM)           // 3072

__device__ __forceinline__ unsigned short f2bf(float f) {
  unsigned int u = __float_as_uint(f);
  u += 0x7FFFu + ((u >> 16) & 1u);            // round-to-nearest-even
  return (unsigned short)(u >> 16);
}

// ---- elementwise fp32 -> bf16 -----------------------------------------------
__global__ void cvt_f32_bf16(const float* __restrict__ in,
                             unsigned short* __restrict__ out, int n) {
  int i = blockIdx.x * blockDim.x + threadIdx.x;
  if (i < n) out[i] = f2bf(in[i]);
}

// ---- tiled transpose + convert: W (K x N fp32) -> WT (N x K bf16) -----------
__global__ void transpose_cvt(const float* __restrict__ W,
                              unsigned short* __restrict__ WT,
                              int K, int N) {
  __shared__ float tile[32][33];
  const int bx = blockIdx.x * 32;   // N
  const int by = blockIdx.y * 32;   // K
  const int tx = threadIdx.x, ty = threadIdx.y;   // block (32,8)
  #pragma unroll
  for (int i = 0; i < 32; i += 8)
    tile[ty + i][tx] = W[(size_t)(by + ty + i) * N + (bx + tx)];
  __syncthreads();
  #pragma unroll
  for (int i = 0; i < 32; i += 8)
    WT[(size_t)(bx + ty + i) * K + (by + tx)] = f2bf(tile[tx][ty + i]);
}

// ---- fragment loaders (wave32, contiguous k = 16*half + s, same for A and B) ---------------------------
// A 16x32 bf16: lane half selects K 0-15 vs 16-31, contiguous; p already
// offset by hsel*16.
__device__ __forceinline__ v16bf ldA(const unsigned short* p, int k) {
  return *reinterpret_cast<const v16bf*>(p + k);
}
// B 32x16 bf16 from B^T: lane = column, lane half selects K 0-15 vs 16-31,
// contiguous in B^T rows; p already offset by hsel*16.
__device__ __forceinline__ v16bf ldB(const unsigned short* p, int k) {
  return *reinterpret_cast<const v16bf*>(p + k);
}

__device__ __forceinline__ v8f wmma_bf16(const v16bf& a, const v16bf& b, const v8f& c) {
  return __builtin_amdgcn_wmma_f32_16x16x32_bf16(
      /*neg_a=*/false, a, /*neg_b=*/false, b,
      /*c_mod=*/(short)0, c, /*reuse_a=*/false, /*reuse_b=*/false);
}

// ---- epilogue: one 16x16 C tile --------------------------------------------
// MODE 0: +b_in then silu (seg 0) / sigmoid (segs 1,2), scatter to 3 buffers
// MODE 1: +bias, store to o0
template <int MODE>
__device__ __forceinline__ void store_tile(const v8f& c, int rowt, int col,
                                           const float* __restrict__ bias,
                                           float* __restrict__ o0,
                                           float* __restrict__ o1,
                                           float* __restrict__ o2) {
  const float bv = bias[col];
  if (MODE == 0) {
    const int seg = col >> 10;                 // 0 inp / 1 gate / 2 forget
    const int cj  = col & (DIM - 1);
    float* dst = (seg == 0) ? o0 : (seg == 1) ? o1 : o2;
    #pragma unroll
    for (int g = 0; g < 8; ++g) {
      const float v = c[g] + bv;
      const float s = 1.0f / (1.0f + __expf(-v));
      dst[(size_t)(rowt + g) * DIM + cj] = (seg == 0) ? v * s : s;
    }
  } else {
    #pragma unroll
    for (int g = 0; g < 8; ++g)
      o0[(size_t)(rowt + g) * DIM + col] = c[g] + bv;
  }
}

// ---- bf16 WMMA GEMM: C(MxN) = A(MxK) * BT(NxK)^T ---------------------------
// Block: 256 threads = 8 waves in a 2(M) x 4(N) grid, 32x32 C per wave,
// 64x128 C per block. KD is compile-time (1024), multiple of 64.
template <int MODE, int KD>
__global__ __launch_bounds__(256) void gemm_wmma_bf16(
    const unsigned short* __restrict__ A,    // M x KD  bf16 row-major
    const unsigned short* __restrict__ BT,   // N x KD  bf16 row-major (B^T)
    const float* __restrict__ bias,          // length N
    float* __restrict__ o0, float* __restrict__ o1, float* __restrict__ o2) {
  const int wave  = threadIdx.x >> 5;
  const int lane  = threadIdx.x & 31;
  const int waveM = wave >> 2;               // 0..1
  const int waveN = wave & 3;                // 0..3
  const int row0  = blockIdx.y * 64 + waveM * 32;
  const int col0  = blockIdx.x * 128 + waveN * 32;
  const int r     = lane & 15;
  const int hsel  = lane >> 4;

  const unsigned short* aP0 = A  + (size_t)(row0 + r) * KD + hsel * 16;
  const unsigned short* aP1 = aP0 + (size_t)16 * KD;
  const unsigned short* bP0 = BT + (size_t)(col0 + r) * KD + hsel * 16;
  const unsigned short* bP1 = bP0 + (size_t)16 * KD;

  v8f c00 = {}, c01 = {}, c10 = {}, c11 = {};

  // ---- software pipeline: ping-pong fragment sets ----
  v16bf a0A = ldA(aP0, 0), a1A = ldA(aP1, 0);
  v16bf b0A = ldB(bP0, 0), b1A = ldB(bP1, 0);
  v16bf a0B, a1B, b0B, b1B;

  int k = 0;
  #pragma unroll 4
  for (; k + 64 < KD; k += 64) {
    // issue next fragment loads (k+32) before consuming set A
    a0B = ldA(aP0, k + 32); a1B = ldA(aP1, k + 32);
    b0B = ldB(bP0, k + 32); b1B = ldB(bP1, k + 32);
    __builtin_prefetch(aP0 + k + 128, 0, 1);   // global_prefetch_b8 (speculative)
    __builtin_prefetch(bP0 + k + 128, 0, 1);
    c00 = wmma_bf16(a0A, b0A, c00);
    c01 = wmma_bf16(a0A, b1A, c01);
    c10 = wmma_bf16(a1A, b0A, c10);
    c11 = wmma_bf16(a1A, b1A, c11);
    // loads for k+64 overlap the WMMAs on set B
    a0A = ldA(aP0, k + 64); a1A = ldA(aP1, k + 64);
    b0A = ldB(bP0, k + 64); b1A = ldB(bP1, k + 64);
    __builtin_prefetch(aP1 + k + 128, 0, 1);
    __builtin_prefetch(bP1 + k + 128, 0, 1);
    c00 = wmma_bf16(a0B, b0B, c00);
    c01 = wmma_bf16(a0B, b1B, c01);
    c10 = wmma_bf16(a1B, b0B, c10);
    c11 = wmma_bf16(a1B, b1B, c11);
  }
  // tail: k == KD-64, two k-steps left (k and k+32)
  a0B = ldA(aP0, k + 32); a1B = ldA(aP1, k + 32);
  b0B = ldB(bP0, k + 32); b1B = ldB(bP1, k + 32);
  c00 = wmma_bf16(a0A, b0A, c00);
  c01 = wmma_bf16(a0A, b1A, c01);
  c10 = wmma_bf16(a1A, b0A, c10);
  c11 = wmma_bf16(a1A, b1A, c11);
  c00 = wmma_bf16(a0B, b0B, c00);
  c01 = wmma_bf16(a0B, b1B, c01);
  c10 = wmma_bf16(a1B, b0B, c10);
  c11 = wmma_bf16(a1B, b1B, c11);

  // C/D layout: VGPR g -> row (g + 8*hsel), lane&15 -> column
  const int rowt = row0 + hsel * 8;
  const int colt = col0 + r;
  store_tile<MODE>(c00, rowt,      colt,      bias, o0, o1, o2);
  store_tile<MODE>(c01, rowt,      colt + 16, bias, o0, o1, o2);
  store_tile<MODE>(c10, rowt + 16, colt,      bias, o0, o1, o2);
  store_tile<MODE>(c11, rowt + 16, colt + 16, bias, o0, o1, o2);
}

// ---- diagonal linear recurrence + gated readout -----------------------------
// one thread per (b,h): state s[d][e'] (2x2), 2048 sequential steps
__global__ void scan_kernel(const float* __restrict__ iAct,
                            const float* __restrict__ gAct,
                            const float* __restrict__ lamB,
                            float* __restrict__ o) {
  const int t = blockIdx.x * blockDim.x + threadIdx.x;   // 0..2047
  const int b = t >> 9;          // BATCH = 4, h in [0,512)
  const int h = t & 511;
  const size_t base0 = (size_t)b * DIM + 2 * h;
  float s00 = 0.f, s01 = 0.f, s10 = 0.f, s11 = 0.f;
  for (int n = 0; n < SEQ; ++n) {
    const size_t base = base0 + (size_t)n * (BATCH * DIM);
    const float2 iv = *reinterpret_cast<const float2*>(iAct + base);
    const float2 gv = *reinterpret_cast<const float2*>(gAct + base);
    const float2 lv = *reinterpret_cast<const float2*>(lamB + base);
    const float d0 = 1.f - lv.x, d1 = 1.f - lv.y;
    s00 = lv.x * s00 + d0 * iv.x;   // s[d=0][e'=0]
    s01 = lv.x * s01 + d0 * iv.y;   // s[0][1]
    s10 = lv.y * s10 + d1 * iv.x;   // s[1][0]
    s11 = lv.y * s11 + d1 * iv.y;   // s[1][1]
    float2 ov;
    ov.x = s00 * gv.x + s10 * gv.y; // o[e'] = sum_d s[d][e'] * gate[d]
    ov.y = s01 * gv.x + s11 * gv.y;
    *reinterpret_cast<float2*>(o + base) = ov;
  }
}

// ---- LayerNorm over d=1024 per row, emit bf16 for GEMM2 ---------------------
__global__ void ln_cvt_kernel(const float* __restrict__ o,
                              const float* __restrict__ gamma,
                              const float* __restrict__ beta,
                              unsigned short* __restrict__ oB) {
  const int row = blockIdx.x;
  const int tid = threadIdx.x;
  const float* x = o + (size_t)row * DIM;
  __shared__ float red[256];
  float s = 0.f;
  for (int j = tid; j < DIM; j += 256) s += x[j];
  red[tid] = s; __syncthreads();
  for (int st = 128; st > 0; st >>= 1) {
    if (tid < st) red[tid] += red[tid + st];
    __syncthreads();
  }
  const float mu = red[0] * (1.0f / DIM);
  __syncthreads();
  float v = 0.f;
  for (int j = tid; j < DIM; j += 256) { const float d = x[j] - mu; v += d * d; }
  red[tid] = v; __syncthreads();
  for (int st = 128; st > 0; st >>= 1) {
    if (tid < st) red[tid] += red[tid + st];
    __syncthreads();
  }
  const float rstd = rsqrtf(red[0] * (1.0f / DIM) + 1e-5f);
  for (int j = tid; j < DIM; j += 256) {
    const float y = (x[j] - mu) * rstd * gamma[j] + beta[j];
    oB[(size_t)row * DIM + j] = f2bf(y);
  }
}

// ---------------------------------------------------------------------------
extern "C" void kernel_launch(void* const* d_in, const int* in_sizes, int n_in,
                              void* d_out, int out_size, void* d_ws, size_t ws_size,
                              hipStream_t stream) {
  const float* x     = (const float*)d_in[0];   // (2048,4,1024)
  const float* W_in  = (const float*)d_in[1];   // (1024,3072)
  const float* b_in  = (const float*)d_in[2];   // (3072)
  const float* W_out = (const float*)d_in[3];   // (1024,1024)
  const float* b_out = (const float*)d_in[4];   // (1024)
  const float* ln_g  = (const float*)d_in[5];   // (1024)
  const float* ln_b  = (const float*)d_in[6];   // (1024)
  float* out = (float*)d_out;                   // (2048,4,1024)

  char* ws = (char*)d_ws;
  unsigned short* xb    = (unsigned short*)(ws);                       // 16 MB
  unsigned short* WinT  = (unsigned short*)(ws + (16u  << 20));        //  6 MB
  unsigned short* WoutT = (unsigned short*)(ws + (22u  << 20));        //  2 MB
  float* iAct = (float*)(ws + (24u  << 20));                           // 32 MB
  float* gAct = (float*)(ws + (56u  << 20));                           // 32 MB
  float* lamB = (float*)(ws + (88u  << 20));                           // 32 MB
  float* oBuf = (float*)(ws + (120u << 20));                           // 32 MB
  unsigned short* oB = (unsigned short*)(ws + (152u << 20));           // 16 MB

  // 1) x -> bf16
  {
    const int n = ROWS * DIM;
    cvt_f32_bf16<<<(n + 255) / 256, 256, 0, stream>>>(x, xb, n);
  }
  // 2) W_in^T, W_out^T -> bf16
  transpose_cvt<<<dim3(N3 / 32, DIM / 32), dim3(32, 8), 0, stream>>>(W_in, WinT, DIM, N3);
  transpose_cvt<<<dim3(DIM / 32, DIM / 32), dim3(32, 8), 0, stream>>>(W_out, WoutT, DIM, DIM);

  // 3) GEMM1: 8192x3072, block tile 64x128 -> grid (24, 128)
  gemm_wmma_bf16<0, DIM><<<dim3(N3 / 128, ROWS / 64), 256, 0, stream>>>(
      xb, WinT, b_in, iAct, gAct, lamB);

  // 4) diagonal scan + gated readout (2048 independent recurrences)
  scan_kernel<<<8, 256, 0, stream>>>(iAct, gAct, lamB, oBuf);

  // 5) LayerNorm -> bf16
  ln_cvt_kernel<<<ROWS, 256, 0, stream>>>(oBuf, ln_g, ln_b, oB);

  // 6) GEMM2: 8192x1024, grid (8, 128), fused b_out -> fp32 out
  gemm_wmma_bf16<1, DIM><<<dim3(DIM / 128, ROWS / 64), 256, 0, stream>>>(
      oB, WoutT, b_out, out, nullptr, nullptr);
}
